// EncoderBlock_84748294685162
// MI455X (gfx1250) — matrix operand order repaired and checked by reading, not compiled, not run on hardware
//
#include <hip/hip_runtime.h>

typedef __attribute__((ext_vector_type(16))) __bf16 v16bf;
typedef __attribute__((ext_vector_type(8)))  float  v8f;
typedef __attribute__((ext_vector_type(4)))  int    v4i;

#define D_MODEL  1024
#define NHEADS   4
#define HEAD_DIM 256
#define DFF      4096
#define SEQ      2048
#define BATCH    4
#define MROWS    (BATCH * SEQ)   // 8192 token rows

#define AS1 __attribute__((address_space(1)))
#define AS3 __attribute__((address_space(3)))

#if defined(__has_builtin)
#if __has_builtin(__builtin_amdgcn_global_load_async_to_lds_b128)
#define USE_ASYNC 1
#else
#define USE_ASYNC 0
#endif
#else
#define USE_ASYNC 0
#endif

#if USE_ASYNC
#warning "CDNA5 async global->LDS path ENABLED"
#else
#warning "CDNA5 async global->LDS path DISABLED (builtin missing; manual copy fallback)"
#endif

// ---------------------------------------------------------------------------
// WMMA helpers (CDNA5 wave32, 16x16x32 bf16 -> f32)
// ---------------------------------------------------------------------------
__device__ __forceinline__ v8f wmma_bf16(v16bf a, v16bf b, v8f c) {
  // (neg_a, A, neg_b, B, c_mod, C, reuse_a, reuse_b)
  return __builtin_amdgcn_wmma_f32_16x16x32_bf16(false, a, false, b, (short)0, c,
                                                 false, false);
}

// A-matrix 16x32 bf16 fragment: lane r=l&15 holds row r;
// VGPR0-3 -> K = 8*hi + 0..7, VGPR4-7 -> K = 16 + 8*hi + 0..7.
__device__ __forceinline__ v16bf load_a_frag(const __bf16* tile, int stride, int lane) {
  const int r = lane & 15, hi = lane >> 4;
  const __bf16* p0 = tile + (size_t)r * stride + hi * 8;
  const __bf16* p1 = p0 + 16;
  v16bf f;
#pragma unroll
  for (int i = 0; i < 8; ++i) { f[i] = p0[i]; f[i + 8] = p1[i]; }
  return f;
}

// B-matrix 32x16 bf16 fragment from a [n][k] tile: lane n=l&15 holds column n,
// K = 8*hi + 0..7, then 16 + 8*hi + 0..7.
__device__ __forceinline__ v16bf load_b_frag(const __bf16* tile, int stride, int lane) {
  const int n = lane & 15, hi = lane >> 4;
  const __bf16* p = tile + (size_t)n * stride + hi * 8;
  v16bf f;
#pragma unroll
  for (int i = 0; i < 8; ++i) { f[i] = p[i]; f[i + 8] = p[i + 16]; }
  return f;
}

// B-matrix 32x16 bf16 fragment from a row-major [k][e] tile (stride between k
// rows): lane n=l&15 -> column e=n, K = 8*hi + i, 16 + 8*hi + i (strided reads).
__device__ __forceinline__ v16bf load_b_frag_strided(const __bf16* tile, int stride,
                                                     int lane) {
  const int n = lane & 15, hi = lane >> 4;
  const __bf16* p = tile + (size_t)hi * 8 * stride + n;
  v16bf f;
#pragma unroll
  for (int i = 0; i < 8; ++i) { f[i] = p[(size_t)i * stride]; f[i + 8] = p[(size_t)(i + 16) * stride]; }
  return f;
}

__device__ __forceinline__ v8f zero8() {
  v8f z = {0.f, 0.f, 0.f, 0.f, 0.f, 0.f, 0.f, 0.f};
  return z;
}

// ---------------------------------------------------------------------------
// Async (or fallback) copy of a 32 x 128 bf16 block, global row stride 256,
// LDS row stride 264 (528 B: 16B-aligned, bank-offset 4 per row).
// 512 16-byte chunks spread over 32 lanes -> 16 async ops per lane.
// ---------------------------------------------------------------------------
__device__ __forceinline__ void copy_block_32x256(const __bf16* __restrict__ g,
                                                  __bf16* l, int lane) {
#pragma unroll
  for (int i = 0; i < 16; ++i) {
    const int chunk = lane + i * 32;          // 0..511
    const int row = chunk >> 4;               // 0..31
    const int c8 = (chunk & 15) * 8;          // bf16 col, 16B granules
    const __bf16* gp = g + (size_t)row * 256 + c8;
    __bf16* lp = l + (size_t)row * 264 + c8;
#if USE_ASYNC
    __builtin_amdgcn_global_load_async_to_lds_b128((AS1 v4i*)gp, (AS3 v4i*)lp, 0, 0);
#else
#pragma unroll
    for (int j = 0; j < 8; ++j) lp[j] = gp[j];
#endif
  }
}

__device__ __forceinline__ void wait_async_all() {
#if USE_ASYNC
#if __has_builtin(__builtin_amdgcn_s_wait_asynccnt)
  __builtin_amdgcn_s_wait_asynccnt(0);
#else
  asm volatile("s_wait_asynccnt 0x0" ::: "memory");
#endif
#endif
}

// ---------------------------------------------------------------------------
// Tiled GEMM: C = A(MxK) * W(KxN) + bias, bf16 WMMA, f32 accumulate.
// Block = 256 threads = 8 waves (4x2), tile 128x128, wave tile 32x64, K step 32.
// QKV mode: blockIdx.z = head h; W -> W + h*K*N; output scattered to [B,H,S,E] bf16.
// ---------------------------------------------------------------------------
template <bool A_BF16, bool OUT_BF16, bool RELU, bool QKV>
__global__ __launch_bounds__(256) void gemm_kernel(const void* __restrict__ Av,
                                                   const float* __restrict__ W,
                                                   const float* __restrict__ bias,
                                                   void* __restrict__ Out,
                                                   int N, int K) {
  __shared__ __bf16 lA[128][40];   // [m][k], padded stride keeps 16B alignment
  __shared__ __bf16 lB[128][40];   // [n][k] (W tile transposed)

  const int t  = threadIdx.x;
  const int n0 = blockIdx.x * 128;
  const int m0 = blockIdx.y * 128;
  const int z  = QKV ? blockIdx.z : 0;
  const float* Wh = W + (size_t)z * K * N;
  const float* bh = bias + (size_t)z * N;

  const int w = t >> 5, lane = t & 31;
  const int wm = w >> 1, wn = w & 1;           // 4 x 2 wave grid
  const int r = lane & 15, hi = lane >> 4;

  v8f acc[2][4];
#pragma unroll
  for (int i = 0; i < 2; ++i)
#pragma unroll
    for (int j = 0; j < 4; ++j) acc[i][j] = zero8();

  const int ar = t >> 1, ac = (t & 1) * 16;    // A staging: 128 rows x 32 cols
  const int wk = t >> 3, wc = (t & 7) * 16;    // W staging: 32 rows x 128 cols

  for (int kk = 0; kk < K; kk += 32) {
    if (A_BF16) {
      const __bf16* ap = (const __bf16*)Av + (size_t)(m0 + ar) * K + kk + ac;
#pragma unroll
      for (int i = 0; i < 16; ++i) lA[ar][ac + i] = ap[i];
      if (kk + 32 < K) __builtin_prefetch(ap + 32, 0, 0);
    } else {
      const float* ap = (const float*)Av + (size_t)(m0 + ar) * K + kk + ac;
#pragma unroll
      for (int i = 0; i < 16; ++i) lA[ar][ac + i] = (__bf16)ap[i];
      if (kk + 32 < K) __builtin_prefetch(ap + 32, 0, 0);
    }
    {
      const float* wp = Wh + (size_t)(kk + wk) * N + n0 + wc;
#pragma unroll
      for (int i = 0; i < 16; ++i) lB[wc + i][wk] = (__bf16)wp[i];
      if (kk + 32 < K) __builtin_prefetch(wp + (size_t)32 * N, 0, 0);
    }
    __syncthreads();

    v16bf a0 = load_a_frag(&lA[wm * 32][0], 40, lane);
    v16bf a1 = load_a_frag(&lA[wm * 32 + 16][0], 40, lane);
#pragma unroll
    for (int j = 0; j < 4; ++j) {
      v16bf bf = load_b_frag(&lB[wn * 64 + j * 16][0], 40, lane);
      acc[0][j] = wmma_bf16(a0, bf, acc[0][j]);
      acc[1][j] = wmma_bf16(a1, bf, acc[1][j]);
    }
    __syncthreads();
  }

  // Epilogue: C layout -> lane holds column (ncol), rows v + 8*hi (+16 for i=1).
#pragma unroll
  for (int j = 0; j < 4; ++j) {
    const int ncol = n0 + wn * 64 + j * 16 + r;
    const float bval = bh[ncol];
#pragma unroll
    for (int i = 0; i < 2; ++i) {
#pragma unroll
      for (int v = 0; v < 8; ++v) {
        float cv = acc[i][j][v] + bval;
        if (RELU) cv = fmaxf(cv, 0.f);
        const int mg = m0 + wm * 32 + i * 16 + v + 8 * hi;
        if (QKV) {
          __bf16* o = (__bf16*)Out;
          const int bb = mg / SEQ, ss = mg - bb * SEQ;
          o[(((size_t)bb * NHEADS + z) * SEQ + ss) * N + ncol] = (__bf16)cv;
        } else if (OUT_BF16) {
          ((__bf16*)Out)[(size_t)mg * N + ncol] = (__bf16)cv;
        } else {
          ((float*)Out)[(size_t)mg * N + ncol] = cv;
        }
      }
    }
  }
}

// ---------------------------------------------------------------------------
// Flash attention. One wave (32 threads) per 16 query rows per (b,h).
// Scores computed transposed (S^T = K_blk * Q^T) so the softmax'd P tile
// converts to an A-operand fragment purely per-lane; kv block = 32.
// K and V blocks double-buffered in LDS via async global->LDS copies.
// ---------------------------------------------------------------------------
__global__ __launch_bounds__(32) void attn_kernel(const __bf16* __restrict__ Q,
                                                  const __bf16* __restrict__ Km,
                                                  const __bf16* __restrict__ Vm,
                                                  float* __restrict__ Z) {
  __shared__ __bf16 lK[2][32][264];   // [buf][kv][e], 33 KB
  __shared__ __bf16 lV[2][32][264];   // [buf][kv][e], 33 KB

  const int lane = threadIdx.x;
  const int bh = blockIdx.y;
  const int b = bh / NHEADS, h = bh % NHEADS;
  const size_t base = (size_t)bh * SEQ * HEAD_DIM;
  const int s0 = blockIdx.x * 16;
  const int r = lane & 15, hi = lane >> 4;

  // Q resident as B-operand fragments (Q^T columns = Q rows), 8 e-chunks of 32.
  v16bf qf[8];
#pragma unroll
  for (int c = 0; c < 8; ++c)
    qf[c] = load_b_frag(Q + base + (size_t)s0 * HEAD_DIM + c * 32, HEAD_DIM, lane);

  v8f oacc[16];   // O (16q x 256e) as 16 C tiles: lane = e, VGPR v -> q = v + 8*hi
#pragma unroll
  for (int et = 0; et < 16; ++et) oacc[et] = zero8();

  float m_cur = -3.0e38f, l_cur = 0.f;
  const float scale = 0.0625f;   // 1/sqrt(256)

  const __bf16* Kbh = Km + base;
  const __bf16* Vbh = Vm + base;

  // Prologue: start copies of block 0 into buffer 0.
  copy_block_32x256(Kbh, &lK[0][0][0], lane);
  copy_block_32x256(Vbh, &lV[0][0][0], lane);

  for (int kb = 0; kb < SEQ; kb += 32) {
    const int p = (kb >> 5) & 1;
    wait_async_all();          // block kb resident in buffer p
    __syncthreads();           // drain prior ds reads of buffer p^1 before reuse
    if (kb + 32 < SEQ) {       // overlap next block's copy with this compute
      copy_block_32x256(Kbh + (size_t)(kb + 32) * HEAD_DIM, &lK[p ^ 1][0][0], lane);
      copy_block_32x256(Vbh + (size_t)(kb + 32) * HEAD_DIM, &lV[p ^ 1][0][0], lane);
    }

    // S^T tiles: C = K_rows(16x32e) x Q^T(32e x 16q); lane = q, VGPR v -> kv.
    v8f sa0 = zero8(), sa1 = zero8();
#pragma unroll
    for (int c = 0; c < 8; ++c) {
      v16bf kf0 = load_a_frag(&lK[p][0][c * 32], 264, lane);
      v16bf kf1 = load_a_frag(&lK[p][16][c * 32], 264, lane);
      sa0 = wmma_bf16(kf0, qf[c], sa0);
      sa1 = wmma_bf16(kf1, qf[c], sa1);
    }

    // Online softmax for query q = lane&15; lane holds kv {8hi+v} of each tile.
    float sm[16];
    float vmax = -3.0e38f;
#pragma unroll
    for (int v = 0; v < 8; ++v) {
      sm[v]     = sa0[v] * scale;
      sm[v + 8] = sa1[v] * scale;
      vmax = fmaxf(vmax, fmaxf(sm[v], sm[v + 8]));
    }
    vmax = fmaxf(vmax, __shfl_xor(vmax, 16, 32));   // combine half-rows
    const float m_new = fmaxf(m_cur, vmax);
    const float alpha = __expf(m_cur - m_new);

    // P as A-operand fragment (16q x 32kv): element i maps per-lane from sm[i].
    v16bf pf;
    float rs = 0.f;
#pragma unroll
    for (int i = 0; i < 16; ++i) {
      const float pv = __expf(sm[i] - m_new);
      rs += pv;
      pf[i] = (__bf16)pv;
    }
    rs += __shfl_xor(rs, 16, 32);
    l_cur = l_cur * alpha + rs;
    m_cur = m_new;

    // Broadcast per-query rescale factors for the O accumulators.
    float a8[8];
#pragma unroll
    for (int v = 0; v < 8; ++v) a8[v] = __shfl(alpha, 8 * hi + v, 32);

#pragma unroll
    for (int et = 0; et < 16; ++et) {
#pragma unroll
      for (int v = 0; v < 8; ++v) oacc[et][v] = oacc[et][v] * a8[v];
      v16bf vf = load_b_frag_strided(&lV[p][0][et * 16], 264, lane);
      oacc[et] = wmma_bf16(pf, vf, oacc[et]);
    }
  }

  // Normalize by row sums and write concat-head output [B,S,D].
  float r8[8];
  const float invl = 1.f / l_cur;
#pragma unroll
  for (int v = 0; v < 8; ++v) r8[v] = __shfl(invl, 8 * hi + v, 32);
#pragma unroll
  for (int et = 0; et < 16; ++et) {
#pragma unroll
    for (int v = 0; v < 8; ++v) {
      const int sg = s0 + v + 8 * hi;
      Z[((size_t)b * SEQ + sg) * D_MODEL + h * HEAD_DIM + et * 16 + r] =
          oacc[et][v] * r8[v];
    }
  }
}

// ---------------------------------------------------------------------------
// out = LayerNorm(x + y) * g + beta, one 256-thread block per row (D=1024).
// ---------------------------------------------------------------------------
__global__ __launch_bounds__(256) void residual_ln_kernel(const float* __restrict__ x,
                                                          const float* __restrict__ y,
                                                          const float* __restrict__ g,
                                                          const float* __restrict__ beta,
                                                          float* __restrict__ out) {
  __shared__ float sred[8], ssred[8];
  const int row = blockIdx.x;
  const int t = threadIdx.x;
  const size_t base = (size_t)row * D_MODEL;

  float v[4];
  float s = 0.f, ss = 0.f;
#pragma unroll
  for (int i = 0; i < 4; ++i) {
    const int c = t * 4 + i;
    v[i] = x[base + c] + y[base + c];
    s += v[i];
    ss += v[i] * v[i];
  }
#pragma unroll
  for (int off = 16; off > 0; off >>= 1) {
    s += __shfl_down(s, off, 32);
    ss += __shfl_down(ss, off, 32);
  }
  if ((t & 31) == 0) { sred[t >> 5] = s; ssred[t >> 5] = ss; }
  __syncthreads();
  float tot = 0.f, tot2 = 0.f;
#pragma unroll
  for (int i = 0; i < 8; ++i) { tot += sred[i]; tot2 += ssred[i]; }
  const float mu = tot * (1.f / D_MODEL);
  const float var = tot2 * (1.f / D_MODEL) - mu * mu;
  const float rstd = rsqrtf(var + 1e-5f);
#pragma unroll
  for (int i = 0; i < 4; ++i) {
    const int c = t * 4 + i;
    out[base + c] = (v[i] - mu) * rstd * g[c] + beta[c];
  }
}

// ---------------------------------------------------------------------------
// Launch pipeline.
// ---------------------------------------------------------------------------
extern "C" void kernel_launch(void* const* d_in, const int* in_sizes, int n_in,
                              void* d_out, int out_size, void* d_ws, size_t ws_size,
                              hipStream_t stream) {
  (void)in_sizes; (void)n_in; (void)out_size; (void)ws_size;

  const float* x   = (const float*)d_in[0];
  const float* Wq  = (const float*)d_in[1];
  const float* bq  = (const float*)d_in[2];
  const float* Wk  = (const float*)d_in[3];
  const float* bk  = (const float*)d_in[4];
  const float* Wv  = (const float*)d_in[5];
  const float* bv  = (const float*)d_in[6];
  const float* Wo  = (const float*)d_in[7];
  const float* bo  = (const float*)d_in[8];
  const float* W1  = (const float*)d_in[9];
  const float* b1  = (const float*)d_in[10];
  const float* W2  = (const float*)d_in[11];
  const float* b2  = (const float*)d_in[12];
  const float* g1  = (const float*)d_in[13];
  const float* be1 = (const float*)d_in[14];
  const float* g2  = (const float*)d_in[15];
  const float* be2 = (const float*)d_in[16];

  char* ws = (char*)d_ws;
  __bf16* Qb = (__bf16*)(ws + ((size_t)0   << 20));   // 16 MB  [B,H,S,E] bf16
  __bf16* Kb = (__bf16*)(ws + ((size_t)16  << 20));   // 16 MB
  __bf16* Vb = (__bf16*)(ws + ((size_t)32  << 20));   // 16 MB
  float*  Zc = (float*)(ws + ((size_t)48  << 20));    // 32 MB  [B,S,D] f32
  float*  Tm = (float*)(ws + ((size_t)80  << 20));    // 32 MB  scratch f32
  float*  H1 = (float*)(ws + ((size_t)112 << 20));    // 32 MB  f32
  __bf16* F1 = (__bf16*)(ws + ((size_t)144 << 20));   // 64 MB  [M,DFF] bf16

  const dim3 blk(256);

  // Q/K/V projections (per head), output bf16 [B,H,S,E].
  const dim3 gq(HEAD_DIM / 128, MROWS / 128, NHEADS);
  gemm_kernel<false, true, false, true><<<gq, blk, 0, stream>>>(x, Wq, bq, Qb,
                                                                HEAD_DIM, D_MODEL);
  gemm_kernel<false, true, false, true><<<gq, blk, 0, stream>>>(x, Wk, bk, Kb,
                                                                HEAD_DIM, D_MODEL);
  gemm_kernel<false, true, false, true><<<gq, blk, 0, stream>>>(x, Wv, bv, Vb,
                                                                HEAD_DIM, D_MODEL);

  // Flash attention -> Zc = concat heads [B,S,D].
  attn_kernel<<<dim3(SEQ / 16, BATCH * NHEADS), dim3(32), 0, stream>>>(Qb, Kb, Vb, Zc);

  // Output projection, residual + LN1.
  gemm_kernel<false, false, false, false><<<dim3(D_MODEL / 128, MROWS / 128, 1), blk, 0,
                                            stream>>>(Zc, Wo, bo, Tm, D_MODEL, D_MODEL);
  residual_ln_kernel<<<MROWS, blk, 0, stream>>>(x, Tm, g1, be1, H1);

  // FFN: relu(H1 @ W1 + b1) (bf16) @ W2 + b2, residual + LN2 -> d_out.
  gemm_kernel<false, true, true, false><<<dim3(DFF / 128, MROWS / 128, 1), blk, 0,
                                          stream>>>(H1, W1, b1, F1, DFF, D_MODEL);
  gemm_kernel<true, false, false, false><<<dim3(D_MODEL / 128, MROWS / 128, 1), blk, 0,
                                           stream>>>(F1, W2, b2, Tm, D_MODEL, DFF);
  residual_ln_kernel<<<MROWS, blk, 0, stream>>>(H1, Tm, g2, be2, (float*)d_out);
}
